// VectorQuantize_31301721653528
// MI455X (gfx1250) — matrix operand order repaired and checked by reading, not compiled, not run on hardware
//
#include <hip/hip_runtime.h>

// ---------------------------------------------------------------------------
// VectorQuantize forward for MI455X (gfx1250, wave32, WMMA)
//   dist(n,k) = |x_n|^2 - 2 x_n.e_k + |e_k|^2 ; argmin_k ; gather ; MSE loss
//   argmin only needs score = e_sq[k] - 2*dot  (|x|^2 is per-row constant)
// GEMM (2*N*K*D = 275 GFLOP) in bf16 WMMA, f32 accumulate.
// Embed chunks streamed to LDS with GLOBAL_LOAD_ASYNC_TO_LDS_B128 (double
// buffered, ASYNCcnt), A fragments hoisted to registers for the whole K loop.
// ---------------------------------------------------------------------------

#define N_PTS 32768
#define DIM   512
#define KCB   8192

#define MT    64        // x rows per block
#define KT    64        // codebook cols per block iteration
#define PITCH 520       // LDS row pitch in bf16 elems (pad vs bank conflicts)

typedef __attribute__((ext_vector_type(16))) __bf16 v16bf;
typedef __attribute__((ext_vector_type(8)))  __bf16 v8bf;
typedef __attribute__((ext_vector_type(4)))  __bf16 v4bf;
typedef __attribute__((ext_vector_type(8)))  float  v8f;
typedef __attribute__((ext_vector_type(4)))  float  v4f;

// workspace layout (bytes)
static constexpr size_t WS_EMBED_BF16 = 0;
static constexpr size_t WS_ESQ        = (size_t)KCB * DIM * 2;        // 8 MB
static constexpr size_t WS_IDX        = WS_ESQ + (size_t)KCB * 4;     // +32 KB
static constexpr size_t WS_LOSS       = WS_IDX + (size_t)N_PTS * 4;   // +128 KB

#define SHUF16(lo, hi) __builtin_shufflevector(lo, hi, \
    0, 1, 2, 3, 4, 5, 6, 7, 8, 9, 10, 11, 12, 13, 14, 15)

// ---------------------------------------------------------------------------
// Kernel 1: embed fp32 -> bf16, e_sq[k] = sum_d embed[k][d]^2, zero loss acc.
// ---------------------------------------------------------------------------
__global__ __launch_bounds__(256) void vq_prep(const float* __restrict__ embed,
                                               __bf16* __restrict__ ebf,
                                               float* __restrict__ esq,
                                               float* __restrict__ loss_acc) {
  if (blockIdx.x == 0 && threadIdx.x == 0) loss_acc[0] = 0.0f;
  const int wid  = threadIdx.x >> 5;
  const int lane = threadIdx.x & 31;
  const int k    = blockIdx.x * 8 + wid;

  const v4f* src = (const v4f*)(embed + (size_t)k * DIM + lane * 16);
  __bf16 tmp[16];
  float sum = 0.0f;
#pragma unroll
  for (int i = 0; i < 4; ++i) {
    v4f f = src[i];
#pragma unroll
    for (int j = 0; j < 4; ++j) {
      sum += f[j] * f[j];
      tmp[4 * i + j] = (__bf16)f[j];
    }
  }
  v8bf* dst = (v8bf*)(ebf + (size_t)k * DIM + lane * 16);
  dst[0] = *(const v8bf*)&tmp[0];
  dst[1] = *(const v8bf*)&tmp[8];

#pragma unroll
  for (int off = 16; off >= 1; off >>= 1) sum += __shfl_xor(sum, off, 32);
  if (lane == 0) esq[k] = sum;
}

// ---------------------------------------------------------------------------
// Kernel 2: WMMA GEMM + running argmin.
// Block: 256 thr = 8 waves.  wave w: wm = w>>1 (M group), wk = w&1 (K half)
// Each wave: 16 rows x 32 cols / iter (2 accumulators); A hoisted in regs.
// Bs double-buffered via global_load_async_to_lds_b128 (ASYNCcnt).
// ---------------------------------------------------------------------------
__global__ __launch_bounds__(256) void vq_argmin(const float* __restrict__ x,
                                                 const __bf16* __restrict__ ebf,
                                                 const float* __restrict__ esq,
                                                 int* __restrict__ out_idx_i,
                                                 float* __restrict__ out_idx_f) {
  __shared__ __align__(16) __bf16 As[MT * PITCH];        // 66560 B
  __shared__ __align__(16) __bf16 Bs[2][KT * PITCH];     // 2 x 66560 B

  const int tid  = threadIdx.x;
  const int lane = tid & 31;
  const int wid  = tid >> 5;
  const int wm   = wid >> 1;        // 0..3  -> rows wm*16..+15
  const int wk   = wid & 1;         // 0..1  -> cols wk*32..+31 of chunk
  const int ln15 = lane & 15;
  const int hi   = lane >> 4;       // 0 or 1 (half-wave)

  // LDS byte base of Bs (generic ptr low 32 bits == LDS byte address)
  const unsigned bsBase = (unsigned)(unsigned long long)(&Bs[0][0]);

  // issue per-lane async memory->LDS copies for one 64x512 bf16 chunk
  auto issue_chunk = [&](int buf, int k0) {
    const __bf16* src = ebf + (size_t)k0 * DIM;
    const unsigned ldsBuf = bsBase + (unsigned)buf * (KT * PITCH * 2);
#pragma unroll
    for (int i = 0; i < 16; ++i) {
      const int e   = tid + i * 256;
      const int row = e >> 6;            // 64 b128 per row
      const int c   = (e & 63) * 8;      // bf16 elem col
      unsigned lds  = ldsBuf + (unsigned)(row * PITCH + c) * 2u;
      unsigned goff = (unsigned)(row * DIM + c) * 2u;
      asm volatile("global_load_async_to_lds_b128 %0, %1, %2"
                   :: "v"(lds), "v"(goff), "s"(src)
                   : "memory");
    }
  };

  // kick off first embed chunk while we stage the x tile
  issue_chunk(0, 0);

  // ---- stage x tile (fp32 -> bf16) into LDS: 64 rows x 512 ----
  const size_t xbase = (size_t)blockIdx.x * MT * DIM;
#pragma unroll 4
  for (int e = tid; e < MT * DIM / 4; e += 256) {
    const int row = e >> 7;            // 128 float4 per row
    const int c   = (e & 127) * 4;
    v4f f = *(const v4f*)(x + xbase + (size_t)row * DIM + c);
    __bf16 h[4] = { (__bf16)f[0], (__bf16)f[1], (__bf16)f[2], (__bf16)f[3] };
    *(v4bf*)(&As[row * PITCH + c]) = *(const v4bf*)h;
  }

  // ---- per-lane fragment addresses (contiguous k = 16*hi + s, same order for A and B) ----
  // A (16x32): lanes 0-15 hold K0-15, lanes 16-31 hold K16-31
  const int aOff  = (wm * 16 + ln15) * PITCH + hi * 16;
  // B (32x16): n = lane&15, lanes 0-15 hold K0-15, lanes 16-31 hold K16-31
  const int bOff0 = (wk * 32 + ln15) * PITCH + hi * 16;
  const int bOff1 = bOff0 + 16 * PITCH;

  asm volatile("s_wait_asynccnt 0" ::: "memory");
  __syncthreads();

  // ---- hoist all 16 A fragments (invariant over the codebook loop) ----
  v16bf a_frag[16];
#pragma unroll
  for (int d = 0; d < 16; ++d) {
    v8bf alo = *(const v8bf*)&As[aOff + d * 32];
    v8bf ahi = *(const v8bf*)&As[aOff + d * 32 + 8];
    a_frag[d] = SHUF16(alo, ahi);
  }

  float bestv[2][8];
  int   besti[2][8];
#pragma unroll
  for (int f = 0; f < 2; ++f)
#pragma unroll
    for (int r = 0; r < 8; ++r) { bestv[f][r] = 3.4e38f; besti[f][r] = 0; }

  // ---- main loop over codebook chunks (double-buffered async Bs) ----
  for (int k0 = 0; k0 < KCB; k0 += KT) {
    const int buf = (k0 / KT) & 1;
    if (k0 + KT < KCB) issue_chunk(buf ^ 1, k0 + KT);   // overlap next load

    const __bf16* bp = &Bs[buf][0];
    v8f acc0 = {};
    v8f acc1 = {};
#pragma unroll
    for (int d = 0; d < 16; ++d) {
      const int dd = d * 32;
      v8bf b0l = *(const v8bf*)&bp[bOff0 + dd];
      v8bf b0h = *(const v8bf*)&bp[bOff0 + dd + 8];
      v16bf b0 = SHUF16(b0l, b0h);
      v8bf b1l = *(const v8bf*)&bp[bOff1 + dd];
      v8bf b1h = *(const v8bf*)&bp[bOff1 + dd + 8];
      v16bf b1 = SHUF16(b1l, b1h);
      acc0 = __builtin_amdgcn_wmma_f32_16x16x32_bf16(
          false, a_frag[d], false, b0, (short)0, acc0, false, false);
      acc1 = __builtin_amdgcn_wmma_f32_16x16x32_bf16(
          false, a_frag[d], false, b1, (short)0, acc1, false, false);
    }

    // score = e_sq - 2*dot ; per-lane running argmin (8 rows per lane)
    const int c0 = k0 + wk * 32 + ln15;
    const int c1 = c0 + 16;
    const float e0 = esq[c0];
    const float e1 = esq[c1];
#pragma unroll
    for (int r = 0; r < 8; ++r) {
      float s0 = e0 - 2.0f * acc0[r];
      if (s0 < bestv[0][r]) { bestv[0][r] = s0; besti[0][r] = c0; }
      float s1 = e1 - 2.0f * acc1[r];
      if (s1 < bestv[1][r]) { bestv[1][r] = s1; besti[1][r] = c1; }
    }

    asm volatile("s_wait_asynccnt 0" ::: "memory");  // next buffer resident
    __syncthreads();                                 // all waves done with cur
  }

  // ---- final reduce: 64 candidates per row, via LDS (alias dead As) ----
  float* redv = (float*)As;                        // 64*64*4 = 16 KB
  int*   redi = (int*)((char*)As + 16384);         // next 16 KB
#pragma unroll
  for (int f = 0; f < 2; ++f) {
    const int e = wk * 32 + f * 16 + ln15;
#pragma unroll
    for (int r = 0; r < 8; ++r) {
      const int m = wm * 16 + r + hi * 8;
      redv[m * 64 + e] = bestv[f][r];
      redi[m * 64 + e] = besti[f][r];
    }
  }
  __syncthreads();

  if (tid < MT) {
    float bv = redv[tid * 64];
    int   bi = redi[tid * 64];
    for (int e = 1; e < 64; ++e) {
      float v  = redv[tid * 64 + e];
      int   i2 = redi[tid * 64 + e];
      if (v < bv || (v == bv && i2 < bi)) { bv = v; bi = i2; }
    }
    const int rowg = blockIdx.x * MT + tid;
    out_idx_i[rowg] = bi;
    out_idx_f[rowg] = (float)bi;
  }
}

// ---------------------------------------------------------------------------
// Kernel 3: gather quantize rows in fp32, accumulate commitment loss.
// ---------------------------------------------------------------------------
__global__ __launch_bounds__(256) void vq_gather(const float* __restrict__ x,
                                                 const float* __restrict__ embed,
                                                 const int* __restrict__ idx,
                                                 float* __restrict__ out_q,
                                                 float* __restrict__ loss_acc) {
  const int wid  = threadIdx.x >> 5;
  const int lane = threadIdx.x & 31;
  const int row  = blockIdx.x * 8 + wid;
  const int k    = idx[row];

  const v4f* ev = (const v4f*)(embed + (size_t)k * DIM + lane * 16);
  const v4f* xv = (const v4f*)(x + (size_t)row * DIM + lane * 16);
  v4f*       qv = (v4f*)(out_q + (size_t)row * DIM + lane * 16);

  float s = 0.0f;
#pragma unroll
  for (int i = 0; i < 4; ++i) {
    v4f e = ev[i];
    v4f xx = xv[i];
    qv[i] = e;                    // forward value == quantize (STE)
#pragma unroll
    for (int j = 0; j < 4; ++j) { float d = e[j] - xx[j]; s += d * d; }
  }
#pragma unroll
  for (int off = 16; off >= 1; off >>= 1) s += __shfl_xor(s, off, 32);
  if (lane == 0) atomicAdd(loss_acc, s);
}

// ---------------------------------------------------------------------------
// Kernel 4: finalize loss = sum / (N*D)   (COMMITMENT_WEIGHT = 1)
// ---------------------------------------------------------------------------
__global__ void vq_finish(const float* __restrict__ loss_acc,
                          float* __restrict__ out_loss) {
  out_loss[0] = loss_acc[0] * (1.0f / ((float)N_PTS * (float)DIM));
}

// ---------------------------------------------------------------------------
extern "C" void kernel_launch(void* const* d_in, const int* in_sizes, int n_in,
                              void* d_out, int out_size, void* d_ws, size_t ws_size,
                              hipStream_t stream) {
  const float* x     = (const float*)d_in[0];   // [N, D] fp32
  const float* embed = (const float*)d_in[1];   // [1, K, D] fp32

  char*   ws       = (char*)d_ws;
  __bf16* ebf      = (__bf16*)(ws + WS_EMBED_BF16);
  float*  esq      = (float*)(ws + WS_ESQ);
  int*    idxbuf   = (int*)(ws + WS_IDX);
  float*  loss_acc = (float*)(ws + WS_LOSS);

  float* out_q     = (float*)d_out;                    // [N, D]
  float* out_idx_f = out_q + (size_t)N_PTS * DIM;      // [N]
  float* out_loss  = out_idx_f + N_PTS;                // [1]

  vq_prep<<<KCB / 8, 256, 0, stream>>>(embed, ebf, esq, loss_acc);
  vq_argmin<<<N_PTS / MT, 256, 0, stream>>>(x, ebf, esq, idxbuf, out_idx_f);
  vq_gather<<<N_PTS / 8, 256, 0, stream>>>(x, embed, idxbuf, out_q, loss_acc);
  vq_finish<<<1, 1, 0, stream>>>(loss_acc, out_loss);
}
